// SelfAttention_52029233824020
// MI455X (gfx1250) — matrix operand order repaired and checked by reading, not compiled, not run on hardware
//
#include <hip/hip_runtime.h>
#include <hip/hip_bf16.h>

// ---------------------------------------------------------------------------
// Self-attention forward, fused flash-attention style, bf16 WMMA + TDM
// (gfx1250).  B=4 T=2048 E=1024 H=16 S=64.
// ---------------------------------------------------------------------------

#define B_ 4
#define T_ 2048
#define E_ 1024
#define H_ 16
#define S_ 64

typedef __attribute__((ext_vector_type(16))) __bf16       v16bf;
typedef __attribute__((ext_vector_type(8)))  float        v8f;
typedef __attribute__((ext_vector_type(4)))  unsigned int u32x4;
typedef __attribute__((ext_vector_type(8)))  int          i32x8;
typedef __attribute__((ext_vector_type(4)))  int          i32x4;

static __device__ inline __bf16 f2bf(float f) { return (__bf16)f; }  // hw cvt

static __device__ inline v8f wmma_bf16(v16bf a, v16bf b, v8f c) {
  return __builtin_amdgcn_wmma_f32_16x16x32_bf16(
      false, a, false, b, (short)0, c, false, false);
}

// ---------------------------------------------------------------------------
// Tensor Data Mover: load a (rows x 64) bf16 tile, rows strided by
// row_stride_elems elements, into LDS.  D# per CDNA5 ISA ch.8.
// Issued once per wave (EXEC ignored); completion via TENSORcnt.
// This toolchain's builtin takes 6 args (g0,g1,g2,g3,g4,cpol).
// ---------------------------------------------------------------------------
static __device__ inline void tdm_load_tile(const void* lds_dst, const void* gsrc,
                                            unsigned rows, unsigned row_stride_elems) {
  unsigned long long ga = (unsigned long long)(size_t)gsrc;
  unsigned la = (unsigned)(size_t)lds_dst;   // generic LDS ptr: [31:0] = LDS byte addr
  u32x4 g0;
  g0[0] = 1u;                                 // count=1, user descriptor
  g0[1] = la;                                 // lds_addr
  g0[2] = (unsigned)ga;                       // global_addr[31:0]
  g0[3] = (unsigned)((ga >> 32) & 0x1FFFFFFull) | (2u << 30);  // addr[56:32]|type=2
  i32x8 g1;
  g1[0] = (int)(1u << 16);                    // data_size=1 -> 2 bytes/elem
  g1[1] = 0;                                  // tensor_dim0 = 1<<20 (low16=0)
  g1[2] = (int)0x00000010u;                   // tensor_dim0 hi=0x10 ; tensor_dim1 lo=0
  g1[3] = (int)((64u << 16) | 0x10u);         // tensor_dim1 hi=0x10 ; tile_dim0=64
  g1[4] = (int)rows;                          // tile_dim1=rows, tile_dim2=0
  g1[5] = (int)row_stride_elems;              // tensor_dim0_stride[31:0]
  g1[6] = 0;                                  // stride hi / dim1_stride lo
  g1[7] = 0;
  i32x4 z4; z4[0] = 0; z4[1] = 0; z4[2] = 0; z4[3] = 0;
  i32x8 z8; z8[0] = 0; z8[1] = 0; z8[2] = 0; z8[3] = 0;
  z8[4] = 0; z8[5] = 0; z8[6] = 0; z8[7] = 0;
  __builtin_amdgcn_tensor_load_to_lds(g0, g1, z4, z4, z8, 0);
}

// --- A fragment (16x32 MxK), row-major bf16 source, two b128 loads ---------
static __device__ inline v16bf load_a_bf(const __bf16* A, int ld, int kofs) {
  int lane = threadIdx.x & 31;
  int m = lane & 15, g = (lane >> 4) & 1;
  const uint4* p = (const uint4*)(A + m * ld + kofs + g * 8);
  union { uint4 q[2]; v16bf v; } u;
  u.q[0] = p[0];   // K = base+0..7
  u.q[1] = p[2];   // K = base+16..23
  return u.v;
}

// Same A fragment from fp32 source (convert on the fly)
static __device__ inline v16bf load_a_f32(const float* A, int ld, int kofs) {
  int lane = threadIdx.x & 31;
  int m = lane & 15, g = (lane >> 4) & 1;
  const float* row = A + m * ld + kofs + g * 8;
  v16bf a;
#pragma unroll
  for (int i = 0; i < 16; ++i) a[i] = f2bf(row[((i >> 3) << 4) + (i & 7)]);
  return a;
}

// --- B fragment (32x16 KxN) where B[K][N] = W[N][K] (multiply by W^T) ------
// Table order (K = 8g+0..7, 16+8g+0..7): two b128 loads per lane.
static __device__ inline v16bf load_b_wT(const __bf16* W, int ld, int n0, int kofs) {
  int lane = threadIdx.x & 31;
  int n = lane & 15, g = (lane >> 4) & 1;
  const uint4* p = (const uint4*)(W + (size_t)(n0 + n) * ld + kofs + g * 8);
  union { uint4 q[2]; v16bf v; } u;
  u.q[0] = p[0];
  u.q[1] = p[2];
  return u.v;
}

// ===========================================================================
// Kernel 1: weight conversion fp32 -> bf16.
// Wq gets 1/E^0.25 * log2(e)  (exp2-domain softmax), Wk gets 1/E^0.25.
// ===========================================================================
__global__ __launch_bounds__(256) void k_convert(
    const float* __restrict__ Wk, const float* __restrict__ Wq,
    const float* __restrict__ Wv, const float* __restrict__ Wp,
    __bf16* wq, __bf16* wk, __bf16* wv, __bf16* wp) {
  int i = blockIdx.x * 256 + threadIdx.x;
  const float sck = 0.17677669529663687f;                 // 1024^-0.25
  const float scq = sck * 1.4426950408889634f;            // * log2(e)
  if (i < S_ * S_) {
    wq[i] = f2bf(Wq[i] * scq);
    wk[i] = f2bf(Wk[i] * sck);
    wv[i] = f2bf(Wv[i]);
  }
  if (i < E_ * E_) wp[i] = f2bf(Wp[i]);
}

// ===========================================================================
// Kernel 2: QKV projection.  One wave per (b,h,16-row t-tile).
// Q,K layout (B,H,T,S);  V stored transposed (B,H,S,T) so the attention
// PV fragments become contiguous b128 LDS loads.
// ===========================================================================
__global__ __launch_bounds__(256) void k_qkv(
    const float* __restrict__ x,
    const __bf16* __restrict__ wq, const __bf16* __restrict__ wk,
    const __bf16* __restrict__ wv,
    __bf16* Q, __bf16* K, __bf16* Vt) {
  const int NT16 = T_ / 16;
  int wave = threadIdx.x >> 5;
  int lane = threadIdx.x & 31;
  int n = lane & 15, g = (lane >> 4) & 1;
  int wid = blockIdx.x * 8 + wave;
  int t0 = (wid % NT16) * 16;
  int h  = (wid / NT16) % H_;
  int b  = wid / (NT16 * H_);

  const float* xrow = x + ((size_t)(b * T_ + t0)) * E_ + h * S_;

  v16bf a[2];
#pragma unroll
  for (int ks = 0; ks < 2; ++ks) a[ks] = load_a_f32(xrow, E_, ks * 32);

  // Q and K: (B,H,T,S)
  const __bf16* Ws[2] = {wq, wk};
  __bf16* Outs[2] = {Q, K};
#pragma unroll
  for (int m = 0; m < 2; ++m) {
    __bf16* obase = Outs[m] + ((size_t)(b * H_ + h) * T_ + t0) * S_;
#pragma unroll
    for (int nt = 0; nt < 4; ++nt) {
      v8f c = {};
#pragma unroll
      for (int ks = 0; ks < 2; ++ks)
        c = wmma_bf16(a[ks], load_b_wT(Ws[m], S_, nt * 16, ks * 32), c);
#pragma unroll
      for (int r = 0; r < 8; ++r)
        obase[(size_t)(r + 8 * g) * S_ + nt * 16 + n] = f2bf(c[r]);
    }
  }
  // V: transposed store into (B,H,S,T)
  {
    __bf16* obase = Vt + ((size_t)(b * H_ + h) * S_) * T_ + t0;
#pragma unroll
    for (int nt = 0; nt < 4; ++nt) {
      v8f c = {};
#pragma unroll
      for (int ks = 0; ks < 2; ++ks)
        c = wmma_bf16(a[ks], load_b_wT(wv, S_, nt * 16, ks * 32), c);
#pragma unroll
      for (int r = 0; r < 8; ++r)
        obase[(size_t)(nt * 16 + n) * T_ + r + 8 * g] = f2bf(c[r]);
    }
  }
}

// ===========================================================================
// Kernel 3: fused flash attention.  One block per (b,h,128-query tile),
// 8 waves x 16 query rows.  K/V tiles staged by the Tensor Data Mover,
// double-buffered; online softmax in exp2 domain.
// ===========================================================================
__global__ __launch_bounds__(256) void k_attn(
    const __bf16* __restrict__ Q, const __bf16* __restrict__ K,
    const __bf16* __restrict__ Vt, __bf16* O) {
  int b = blockIdx.z, h = blockIdx.y, q0 = blockIdx.x * 128;
  int wave = threadIdx.x >> 5;
  int lane = threadIdx.x & 31;
  int n = lane & 15, g = (lane >> 4) & 1;

  __shared__ __align__(16) __bf16 Qs[128 * 64];      // 16 KB
  __shared__ __align__(16) __bf16 Ks[2][64 * 64];    // 16 KB
  __shared__ __align__(16) __bf16 Vs[2][64 * 64];    // 16 KB (rows=s, cols=t)
  __shared__ __align__(16) __bf16 Ps[8][16 * 64];    // 16 KB

  const __bf16* Qg = Q  + ((size_t)(b * H_ + h) * T_ + q0) * S_;
  const __bf16* Kg = K  + ((size_t)(b * H_ + h) * T_) * S_;
  const __bf16* Vg = Vt + ((size_t)(b * H_ + h) * S_) * T_;

  // preload: Q tile + first K/V tiles via TDM (wave 0 only)
  if (wave == 0) {
    tdm_load_tile(Qs,    Qg, 128, S_);
    tdm_load_tile(Ks[0], Kg, 64,  S_);
    tdm_load_tile(Vs[0], Vg, 64,  T_);
    __builtin_amdgcn_s_wait_tensorcnt(0);
  }
  __syncthreads();

  v8f oacc[4] = {};
  float mrun[8], lrun[8];
#pragma unroll
  for (int r = 0; r < 8; ++r) { mrun[r] = -1e30f; lrun[r] = 0.f; }

  const __bf16* Qw = Qs + wave * 16 * 64;
  __bf16* Pw = &Ps[wave][0];

  const int NKT = T_ / 64;
  for (int kt = 0; kt < NKT; ++kt) {
    int cur = kt & 1, nxt = cur ^ 1;
    // prefetch next K/V tiles into the other buffer while we compute
    if (wave == 0 && kt + 1 < NKT) {
      tdm_load_tile(Ks[nxt], Kg + (size_t)(kt + 1) * 64 * S_, 64, S_);
      tdm_load_tile(Vs[nxt], Vg + (size_t)(kt + 1) * 64,      64, T_);
    }

    // S = Q * K^T (16x64 per wave), logits already in log2 domain
    v8f sacc[4] = {};
#pragma unroll
    for (int nt = 0; nt < 4; ++nt)
#pragma unroll
      for (int ks = 0; ks < 2; ++ks)
        sacc[nt] = wmma_bf16(load_a_bf(Qw, 64, ks * 32),
                             load_b_wT(Ks[cur], 64, nt * 16, ks * 32), sacc[nt]);

    // online softmax; C layout: lane holds rows r+8g, col n
    float rowm[8];
#pragma unroll
    for (int r = 0; r < 8; ++r)
      rowm[r] = fmaxf(fmaxf(sacc[0][r], sacc[1][r]), fmaxf(sacc[2][r], sacc[3][r]));
#pragma unroll
    for (int ofs = 1; ofs < 16; ofs <<= 1)
#pragma unroll
      for (int r = 0; r < 8; ++r)
        rowm[r] = fmaxf(rowm[r], __shfl_xor(rowm[r], ofs, 32));

    float corr[8];
#pragma unroll
    for (int r = 0; r < 8; ++r) {
      float mn = fmaxf(mrun[r], rowm[r]);
      corr[r] = exp2f(mrun[r] - mn);
      mrun[r] = mn;
    }

    float rs[8];
#pragma unroll
    for (int r = 0; r < 8; ++r) rs[r] = 0.f;
#pragma unroll
    for (int nt = 0; nt < 4; ++nt)
#pragma unroll
      for (int r = 0; r < 8; ++r) {
        float p = exp2f(sacc[nt][r] - mrun[r]);
        sacc[nt][r] = p;
        rs[r] += p;
      }
#pragma unroll
    for (int ofs = 1; ofs < 16; ofs <<= 1)
#pragma unroll
      for (int r = 0; r < 8; ++r) rs[r] += __shfl_xor(rs[r], ofs, 32);
#pragma unroll
    for (int r = 0; r < 8; ++r) lrun[r] = lrun[r] * corr[r] + rs[r];
#pragma unroll
    for (int nt = 0; nt < 4; ++nt)
#pragma unroll
      for (int r = 0; r < 8; ++r) oacc[nt][r] *= corr[r];

    // P: C layout -> per-wave LDS -> A layout (DS in-order within wave)
#pragma unroll
    for (int nt = 0; nt < 4; ++nt)
#pragma unroll
      for (int r = 0; r < 8; ++r)
        Pw[(r + 8 * g) * 64 + nt * 16 + n] = f2bf(sacc[nt][r]);

    // O += P * V   (Vs rows = s, so B[K=t][N=s] = Vs[s][t] -> wT loader)
#pragma unroll
    for (int nt = 0; nt < 4; ++nt)
#pragma unroll
      for (int ks = 0; ks < 2; ++ks)
        oacc[nt] = wmma_bf16(load_a_bf(Pw, 64, ks * 32),
                             load_b_wT(Vs[cur], 64, nt * 16, ks * 32), oacc[nt]);

    if (wave == 0) __builtin_amdgcn_s_wait_tensorcnt(0);
    __syncthreads();   // publish next buffers / protect cur buffers
  }

  // normalize, store O in (B,T,H,S) bf16
  float inv[8];
#pragma unroll
  for (int r = 0; r < 8; ++r) inv[r] = 1.f / lrun[r];
#pragma unroll
  for (int nt = 0; nt < 4; ++nt)
#pragma unroll
    for (int r = 0; r < 8; ++r) {
      int trow = q0 + wave * 16 + r + 8 * g;
      O[(((size_t)b * T_ + trow) * H_ + h) * S_ + nt * 16 + n] =
          f2bf(oacc[nt][r] * inv[r]);
    }
}

// ===========================================================================
// Kernel 4: output projection out = O(8192x1024) * Wp^T + bp (fp32 out).
// 128x64 output tile per block (8 waves), TDM-staged, double-buffered K loop.
// ===========================================================================
__global__ __launch_bounds__(256) void k_proj(
    const __bf16* __restrict__ Ob, const __bf16* __restrict__ wp,
    const float* __restrict__ bp, float* __restrict__ out) {
  int m0 = blockIdx.x * 128;
  int n0 = blockIdx.y * 64;
  int wave = threadIdx.x >> 5;
  int lane = threadIdx.x & 31;
  int n = lane & 15, g = (lane >> 4) & 1;

  __shared__ __align__(16) __bf16 As[2][128 * 64];   // 32 KB
  __shared__ __align__(16) __bf16 Bs[2][64 * 64];    // 16 KB

  if (wave == 0) {
    tdm_load_tile(As[0], Ob + (size_t)m0 * E_, 128, E_);
    tdm_load_tile(Bs[0], wp + (size_t)n0 * E_, 64, E_);
    __builtin_amdgcn_s_wait_tensorcnt(0);
  }
  __syncthreads();

  v8f acc[4] = {};
  const int NKB = E_ / 64;
  for (int kb = 0; kb < NKB; ++kb) {
    int cur = kb & 1, nxt = cur ^ 1;
    if (wave == 0 && kb + 1 < NKB) {
      tdm_load_tile(As[nxt], Ob + (size_t)m0 * E_ + (kb + 1) * 64, 128, E_);
      tdm_load_tile(Bs[nxt], wp + (size_t)n0 * E_ + (kb + 1) * 64, 64, E_);
    }
    const __bf16* Aw = As[cur] + wave * 16 * 64;
#pragma unroll
    for (int nt = 0; nt < 4; ++nt)
#pragma unroll
      for (int ks = 0; ks < 2; ++ks)
        acc[nt] = wmma_bf16(load_a_bf(Aw, 64, ks * 32),
                            load_b_wT(Bs[cur], 64, nt * 16, ks * 32), acc[nt]);
    if (wave == 0) __builtin_amdgcn_s_wait_tensorcnt(0);
    __syncthreads();
  }

#pragma unroll
  for (int nt = 0; nt < 4; ++nt)
#pragma unroll
    for (int r = 0; r < 8; ++r) {
      int row = m0 + wave * 16 + r + 8 * g;
      int col = n0 + nt * 16 + n;
      out[(size_t)row * E_ + col] = acc[nt][r] + bp[col];
    }
}

// ===========================================================================
extern "C" void kernel_launch(void* const* d_in, const int* in_sizes, int n_in,
                              void* d_out, int out_size, void* d_ws, size_t ws_size,
                              hipStream_t stream) {
  const float* x  = (const float*)d_in[0];
  const float* Wk = (const float*)d_in[1];
  const float* Wq = (const float*)d_in[2];
  const float* Wv = (const float*)d_in[3];
  const float* Wp = (const float*)d_in[4];
  const float* bp = (const float*)d_in[5];
  float* out = (float*)d_out;

  char* ws = (char*)d_ws;
  const size_t oWq = 0;
  const size_t oWk = oWq + (size_t)S_ * S_ * 2;
  const size_t oWv = oWk + (size_t)S_ * S_ * 2;
  const size_t oWp = oWv + (size_t)S_ * S_ * 2;
  const size_t oQ  = oWp + (size_t)E_ * E_ * 2;
  const size_t oK  = oQ + (size_t)B_ * T_ * E_ * 2;
  const size_t oV  = oK + (size_t)B_ * T_ * E_ * 2;
  const size_t oO  = oV + (size_t)B_ * T_ * E_ * 2;

  __bf16* wq = (__bf16*)(ws + oWq);
  __bf16* wk = (__bf16*)(ws + oWk);
  __bf16* wv = (__bf16*)(ws + oWv);
  __bf16* wp = (__bf16*)(ws + oWp);
  __bf16* Qb = (__bf16*)(ws + oQ);
  __bf16* Kb = (__bf16*)(ws + oK);
  __bf16* Vb = (__bf16*)(ws + oV);   // transposed (B,H,S,T)
  __bf16* Ob = (__bf16*)(ws + oO);

  k_convert<<<(E_ * E_) / 256, 256, 0, stream>>>(Wk, Wq, Wv, Wp, wq, wk, wv, wp);
  k_qkv<<<(B_ * H_ * (T_ / 16)) / 8, 256, 0, stream>>>(x, wq, wk, wv, Qb, Kb, Vb);
  k_attn<<<dim3(T_ / 128, H_, B_), 256, 0, stream>>>(Qb, Kb, Vb, Ob);
  k_proj<<<dim3((B_ * T_) / 128, E_ / 64), 256, 0, stream>>>(Ob, wp, bp, out);
}
